// LatticeRNN_31121333027060
// MI455X (gfx1250) — matrix operand order repaired and checked by reading, not compiled, not run on hardware
//
#include <hip/hip_runtime.h>
#include <stdint.h>

// Problem constants (B,S,D,H,K) = (32,512,512,512,4), gates = 4H = 2048.
#define B_    32
#define S_    512
#define D_    512
#define H_    512
#define G4    2048
#define KARC  4

typedef __attribute__((ext_vector_type(16))) __bf16       v16bf;
typedef __attribute__((ext_vector_type(8)))  float        v8f;
typedef __attribute__((ext_vector_type(4)))  unsigned int u32x4;

union FragU {           // 32 bytes: 16 bf16 lane-elements of a WMMA A/B fragment
    u32x4 q[2];
    v16bf v;
};

__device__ __forceinline__ unsigned short f32_to_bf16(float f) {
    union { float f; unsigned int u; } c; c.f = f;
    unsigned int u = c.u;
    unsigned int r = u + 0x7FFFu + ((u >> 16) & 1u);   // round-to-nearest-even
    if (((u >> 23) & 0xFFu) == 0xFFu) r = u;           // keep NaN/Inf payload
    return (unsigned short)(r >> 16);
}

// ---------------------------------------------------------------------------
// K0: convert W_ih / W_hh to bf16 once; reset the global barrier counter.
// ---------------------------------------------------------------------------
__global__ __launch_bounds__(256) void prep_kernel(
    const float* __restrict__ Wih, const float* __restrict__ Whh,
    unsigned short* __restrict__ WihB, unsigned short* __restrict__ WhhB,
    unsigned int* __restrict__ bar)
{
    int i = blockIdx.x * 256 + threadIdx.x;            // grid covers G4*D exactly
    WihB[i] = f32_to_bf16(Wih[i]);
    WhhB[i] = f32_to_bf16(Whh[i]);
    if (i == 0)
        __hip_atomic_store(bar, 0u, __ATOMIC_RELAXED, __HIP_MEMORY_SCOPE_AGENT);
}

// ---------------------------------------------------------------------------
// K1: gates_x[r = t*32 + b, n] = x[b,t,:] @ W_ih^T + (b_ih + b_hh)   (WMMA bf16)
// One block = one timestep t (32 rows); 8 waves x (16 ntiles x 2 mtiles) tiles.
// ---------------------------------------------------------------------------
__global__ __launch_bounds__(256) void gatesx_kernel(
    const float* __restrict__ x,                 // [B,S,D]
    const unsigned short* __restrict__ WihB,     // [G4,D] bf16 bits
    const float* __restrict__ bih, const float* __restrict__ bhh,
    float* __restrict__ gatesX)                  // [S*B, G4]
{
    __shared__ unsigned short Als[32 * 520];     // 32 rows, padded stride
    const int tid = threadIdx.x;
    const int r0  = blockIdx.x * 32;             // t = blockIdx.x, b = row idx

    for (int idx = tid; idx < 32 * 512; idx += 256) {
        int m = idx >> 9, k = idx & 511;
        int r = r0 + m, t = r >> 5, b = r & 31;
        Als[m * 520 + k] = f32_to_bf16(x[((size_t)b * S_ + t) * D_ + k]);
    }
    __syncthreads();

    const int w = tid >> 5, l = tid & 31;
    const int lh = l >> 4, ln = l & 15;
    for (int i = 0; i < 16; ++i) {
        int nt = w * 16 + i;
        int n  = nt * 16 + ln;
        float bb = bih[n] + bhh[n];
        v8f acc[2];
        #pragma unroll
        for (int mt = 0; mt < 2; ++mt)
            #pragma unroll
            for (int v = 0; v < 8; ++v) acc[mt][v] = bb;

        const unsigned short* brow = WihB + (size_t)n * D_;
        for (int kb = 0; kb < 16; ++kb) {
            FragU bf;
            int k0 = kb * 32 + lh * 16;          // B: lanes<16 K=0..15, >=16 K=16..31
            bf.q[0] = *(const u32x4*)(brow + k0);
            bf.q[1] = *(const u32x4*)(brow + k0 + 8);
            #pragma unroll
            for (int mt = 0; mt < 2; ++mt) {
                FragU af;
                int m = mt * 16 + ln;            // A: lane%16 -> row M
                const unsigned short* ar = Als + m * 520 + kb * 32 + lh * 16;
                af.q[0] = *(const u32x4*)(ar);       // K 0.. 7 (+16 hi half)
                af.q[1] = *(const u32x4*)(ar + 8);   // K 8..15 (+16 hi half)
                acc[mt] = __builtin_amdgcn_wmma_f32_16x16x32_bf16(
                    false, af.v, false, bf.v, (short)0, acc[mt], false, false);
            }
        }
        #pragma unroll
        for (int mt = 0; mt < 2; ++mt)
            #pragma unroll
            for (int v = 0; v < 8; ++v) {
                int m = mt * 16 + v + 8 * lh;
                gatesX[(size_t)(r0 + m) * G4 + n] = acc[mt][v];
            }
    }
}

// ---------------------------------------------------------------------------
// K2: persistent sequential lattice-LSTM. 32 blocks, block owns 16 h-columns.
// The block's 64-row W_hh slice (4 gates x 16 cols x 512 k, bf16 = 64KB) is
// preloaded into LDS once, so the per-step GEMM touches only LDS + gatesX.
// Per step: gather h_prev/c_prev, 8 gate tiles via WMMA, LSTM update,
// one device-scope barrier.
// ---------------------------------------------------------------------------
__global__ __launch_bounds__(256) void seq_kernel(
    const float* __restrict__ prev_w,            // [B,S,K]
    const int*   __restrict__ prev_idx,          // [B,S,K]
    const unsigned short* __restrict__ WhhB,     // [G4,H] bf16 bits
    const float* __restrict__ gatesX,            // [S*B, G4]
    float* __restrict__ hOut,                    // d_out [B,S,H] == h history
    float* __restrict__ cHist,                   // [B,S,H]
    unsigned int* __restrict__ bar)
{
    __shared__ unsigned short Wls[64 * 520];     // block's W_hh slice (bf16)
    __shared__ unsigned short hprev[B_ * 520];   // bf16 A-operand, padded rows
    __shared__ float cprev[B_ * 16];
    __shared__ float gtile[4][B_][16];           // i,f,g,o tiles for this slice
    __shared__ float wls[B_][KARC];
    __shared__ int   ils[B_][KARC];

    const int tid = threadIdx.x;
    const int blk = blockIdx.x;                  // 0..31
    const int hc0 = blk * 16;
    const int w = tid >> 5, l = tid & 31;
    const int lh = l >> 4, ln = l & 15;
    const int mt = w & 1, gate = w >> 1;
    const int nbase = gate * H_ + hc0;

    // one-time W_hh slice preload: local row wr = gate*16 + col
    for (int idx = tid; idx < 64 * 512; idx += 256) {
        int wr = idx >> 9, k = idx & 511;
        int g = wr >> 4, rn = wr & 15;
        Wls[wr * 520 + k] = WhhB[(size_t)(g * H_ + hc0 + rn) * H_ + k];
    }
    const unsigned short* browL = Wls + (gate * 16 + ln) * 520;

    for (int t = 0; t < S_; ++t) {
        // --- stage lattice arcs for this step ---
        if (tid < B_ * KARC) {
            int b = tid >> 2, j = tid & 3;
            size_t o = ((size_t)b * S_ + t) * KARC + j;
            wls[b][j] = prev_w[o];
            ils[b][j] = prev_idx[o];
        }
        __syncthreads();   // also covers the one-time Wls preload at t==0

        // --- build h_prev (bf16) and c_prev (f32) in LDS ---
        if (t == 0) {
            for (int idx = tid; idx < B_ * 512; idx += 256) {
                int b = idx >> 9, k = idx & 511;
                hprev[b * 520 + k] = 0;
            }
            for (int e = tid; e < B_ * 16; e += 256) cprev[e] = 0.0f;
        } else {
            int k = tid * 2;                     // 2 consecutive columns / thread
            for (int b = 0; b < B_; ++b) {
                float a0 = 0.f, a1 = 0.f;
                #pragma unroll
                for (int j = 0; j < KARC; ++j) {
                    float wj = wls[b][j];
                    float2 hv = *(const float2*)(hOut +
                        ((size_t)b * S_ + ils[b][j]) * H_ + k);
                    a0 += wj * hv.x; a1 += wj * hv.y;
                }
                unsigned int packed = (unsigned int)f32_to_bf16(a0) |
                                      ((unsigned int)f32_to_bf16(a1) << 16);
                *(unsigned int*)(hprev + b * 520 + k) = packed;
            }
            #pragma unroll
            for (int q = 0; q < 2; ++q) {
                int e = tid * 2 + q;
                int b = e >> 4, hl = e & 15;
                float a = 0.f;
                #pragma unroll
                for (int j = 0; j < KARC; ++j)
                    a += wls[b][j] *
                         cHist[((size_t)b * S_ + ils[b][j]) * H_ + hc0 + hl];
                cprev[e] = a;
            }
        }
        __syncthreads();

        // --- one 16x16 gate tile per wave: gates_x + h_prev @ W_hh^T ---
        v8f acc;
        #pragma unroll
        for (int v = 0; v < 8; ++v) {
            int m = mt * 16 + v + 8 * lh;        // batch row 0..31
            acc[v] = gatesX[((size_t)t * B_ + m) * G4 + nbase + ln];
        }
        for (int kb = 0; kb < 16; ++kb) {
            FragU bf, af;
            int k0 = kb * 32 + lh * 16;
            bf.q[0] = *(const u32x4*)(browL + k0);
            bf.q[1] = *(const u32x4*)(browL + k0 + 8);
            int m = mt * 16 + ln;
            const unsigned short* ar = hprev + m * 520 + kb * 32 + lh * 16;
            af.q[0] = *(const u32x4*)(ar);
            af.q[1] = *(const u32x4*)(ar + 8);
            acc = __builtin_amdgcn_wmma_f32_16x16x32_bf16(
                false, af.v, false, bf.v, (short)0, acc, false, false);
        }
        #pragma unroll
        for (int v = 0; v < 8; ++v) {
            int m = mt * 16 + v + 8 * lh;
            gtile[gate][m][ln] = acc[v];
        }
        __syncthreads();

        // --- LSTM elementwise for this block's 32x16 slice ---
        #pragma unroll
        for (int q = 0; q < 2; ++q) {
            int e = tid + q * 256;
            int b = e >> 4, hl = e & 15;
            float iv = gtile[0][b][hl], fv = gtile[1][b][hl];
            float gv = gtile[2][b][hl], ov = gtile[3][b][hl];
            float si = 1.f / (1.f + __expf(-iv));
            float sf = 1.f / (1.f + __expf(-fv));
            float so = 1.f / (1.f + __expf(-ov));
            float cn = sf * cprev[e] + si * tanhf(gv);
            float hn = so * tanhf(cn);
            size_t o = ((size_t)b * S_ + t) * H_ + hc0 + hl;
            hOut[o]  = hn;
            cHist[o] = cn;
        }

        // --- device-scope epoch barrier (release stores, then sync 32 blocks)
        __threadfence();
        __syncthreads();
        if (tid == 0) {
            __hip_atomic_fetch_add(bar, 1u, __ATOMIC_ACQ_REL,
                                   __HIP_MEMORY_SCOPE_AGENT);
            unsigned int target = 32u * (unsigned int)(t + 1);
            while (__hip_atomic_load(bar, __ATOMIC_ACQUIRE,
                                     __HIP_MEMORY_SCOPE_AGENT) < target) {
                __builtin_amdgcn_s_sleep(1);
            }
        }
        __syncthreads();   // also separates LDS reuse across steps
    }
}

// ---------------------------------------------------------------------------
// Workspace layout (requires ~172 MB):
//   [0)              gates_x  f32 [S*B, 4H]   = 134217728 B
//   [134217728)      c_hist   f32 [B,S,H]     =  33554432 B
//   [167772160)      W_ih bf16 [4H,D]         =   2097152 B
//   [169869312)      W_hh bf16 [4H,H]         =   2097152 B
//   [171966464)      barrier counter (u32)
// ---------------------------------------------------------------------------
extern "C" void kernel_launch(void* const* d_in, const int* in_sizes, int n_in,
                              void* d_out, int out_size, void* d_ws, size_t ws_size,
                              hipStream_t stream) {
    const float* x      = (const float*)d_in[0];
    const float* prev_w = (const float*)d_in[1];
    const float* Wih    = (const float*)d_in[2];
    const float* Whh    = (const float*)d_in[3];
    const float* bih    = (const float*)d_in[4];
    const float* bhh    = (const float*)d_in[5];
    const int*   pidx   = (const int*)d_in[6];
    float* hOut = (float*)d_out;

    char* ws = (char*)d_ws;
    float*          gatesX = (float*)(ws);
    float*          cHist  = (float*)(ws + 134217728);
    unsigned short* WihB   = (unsigned short*)(ws + 167772160);
    unsigned short* WhhB   = (unsigned short*)(ws + 169869312);
    unsigned int*   bar    = (unsigned int*)(ws + 171966464);

    prep_kernel  <<<4096, 256, 0, stream>>>(Wih, Whh, WihB, WhhB, bar);
    gatesx_kernel<<< 512, 256, 0, stream>>>(x, WihB, bih, bhh, gatesX);
    seq_kernel   <<<  32, 256, 0, stream>>>(prev_w, pidx, WhhB, gatesX,
                                            hOut, cHist, bar);
}
